// UncertaintyEstimation_77567109366451
// MI455X (gfx1250) — matrix operand order repaired and checked by reading, not compiled, not run on hardware
//
#include <hip/hip_runtime.h>
#include <hip/hip_bf16.h>

// ---------------------------------------------------------------------------
// UncertaintyEstimation on MI455X (gfx1250, wave32, WMMA bf16 16x16x32)
//
//   B=4, C=64, C2=128, H=W=256, T=16
//   stage 1: 3x3 conv C->C2 (implicit GEMM, WMMA bf16)     ~38.7 GFLOP
//   stage 2: [64,128] masked-sample GEMM + tanh + mean/var  ~4.3 GFLOP
//            + 3x3 conv C2->4 (AU) + sigmoid                ~2.4 GFLOP
//   Bandwidth-bound (~250 MB @ 23.3 TB/s); h kept in bf16 channels-last.
// ---------------------------------------------------------------------------

typedef __bf16 bf16_t;
typedef __attribute__((ext_vector_type(16))) __bf16 v16bf;
typedef __attribute__((ext_vector_type(8)))  __bf16 v8bf;
typedef __attribute__((ext_vector_type(8)))  float  v8f;

#define B_   4
#define C_   64
#define C2_  128
#define H_   256
#define W_   256
#define T_   16
#define HP_  (H_ + 2)
#define WP_  (W_ + 2)

union U16 { v16bf v; v8bf h[2]; };

// A-matrix fragment (16x32 bf16): lane l<16 holds row l, K = kb+[0..15]
// contiguous; lanes 16..31 hold the same rows with K = kb+16+[0..15].
__device__ __forceinline__ v16bf load_a_frag(const bf16_t* __restrict__ row,
                                             int kb, int half) {
  U16 u;
  u.h[0] = *(const v8bf*)(row + kb + half * 16);
  u.h[1] = *(const v8bf*)(row + kb + half * 16 + 8);
  return u.v;
}

// B-matrix fragment (32x16 bf16): lane l<16 holds col l, K = kb+[0..15]
// contiguous; lanes 16..31 hold K = kb+16+[0..15].  One 32-byte load.
__device__ __forceinline__ v16bf load_b_frag(const bf16_t* __restrict__ pix,
                                             int kb, int half) {
  return *(const v16bf*)(pix + kb + half * 16);
}

// ---------------------------------------------------------------------------
// zero padded activation buffers (borders must be 0 for implicit padding)
// ---------------------------------------------------------------------------
__global__ void zero_kernel(uint4* __restrict__ p, size_t n16) {
  size_t i = (size_t)blockIdx.x * blockDim.x + threadIdx.x;
  size_t stride = (size_t)gridDim.x * blockDim.x;
  uint4 z = {0u, 0u, 0u, 0u};
  for (; i < n16; i += stride) p[i] = z;
}

// ---------------------------------------------------------------------------
// x [B,C,H,W] f32  ->  x_pad [B,HP,WP,C] bf16 (channels-last, zero halo)
// ---------------------------------------------------------------------------
__global__ void pack_x_kernel(const float* __restrict__ x,
                              bf16_t* __restrict__ x_pad) {
  size_t idx = (size_t)blockIdx.x * blockDim.x + threadIdx.x;
  const size_t npix = (size_t)B_ * H_ * W_;
  if (idx >= npix) return;
  int b  = (int)(idx / ((size_t)H_ * W_));
  int rp = (int)(idx % ((size_t)H_ * W_));
  int y  = rp / W_;
  int xw = rp % W_;
  const float* src = x + (size_t)b * C_ * H_ * W_ + rp;  // c stride = H*W
  bf16_t* dst = x_pad + (((size_t)b * HP_ + (y + 1)) * WP_ + (xw + 1)) * C_;
  for (int c0 = 0; c0 < C_; c0 += 8) {
    v8bf t;
    for (int j = 0; j < 8; ++j)
      t[j] = (bf16_t)src[(size_t)(c0 + j) * H_ * W_];
    *(v8bf*)(dst + c0) = t;
  }
}

// ---------------------------------------------------------------------------
// weight packing:
//   Wc   [9][C2][C]  bf16: conv1 weights per tap, A-major (oc rows, ic contig)
//   Wall [64][C2]    bf16: (W_out[o,:] * masks[t,:]) at row t*4+o
//   Waue [9][16][C2] bf16: AU conv weights per tap, rows 4..15 zero
// ---------------------------------------------------------------------------
__global__ void prep_w_kernel(const float* __restrict__ Wconv,
                              const float* __restrict__ Wout,
                              const float* __restrict__ masks,
                              const float* __restrict__ Waue_src,
                              bf16_t* __restrict__ Wc,
                              bf16_t* __restrict__ Wall,
                              bf16_t* __restrict__ Waue) {
  const int nWc   = 9 * C2_ * C_;   // 73728
  const int nWall = 64 * C2_;       // 8192
  const int nWaue = 9 * 16 * C2_;   // 18432
  int idx = blockIdx.x * blockDim.x + threadIdx.x;
  if (idx < nWc) {
    int tap = idx / (C2_ * C_);
    int r   = idx % (C2_ * C_);
    int oc = r / C_, ic = r % C_;
    int ty = tap / 3, tx = tap % 3;
    Wc[idx] = (bf16_t)Wconv[(((size_t)oc * C_ + ic) * 3 + ty) * 3 + tx];
  } else if (idx < nWc + nWall) {
    int i = idx - nWc;
    int row = i / C2_, c = i % C2_;
    int t = row >> 2, o = row & 3;
    Wall[i] = (bf16_t)(Wout[o * C2_ + c] * masks[t * C2_ + c]);
  } else if (idx < nWc + nWall + nWaue) {
    int i = idx - nWc - nWall;
    int tap = i / (16 * C2_);
    int r2  = i % (16 * C2_);
    int row = r2 / C2_, c = r2 % C2_;
    int ty = tap / 3, tx = tap % 3;
    Waue[i] = (row < 4)
        ? (bf16_t)Waue_src[(((size_t)row * C2_ + c) * 3 + ty) * 3 + tx]
        : (bf16_t)0.0f;
  }
}

// ---------------------------------------------------------------------------
// stage 1: implicit-GEMM 3x3 conv 64->128, WMMA bf16, f32 accum.
// Block = 256 threads = 8 waves. Each wave: 16 out-channels (tile = wave id)
// x 16 consecutive pixels of one row; K-loop = 9 taps x 2 (ic 0..31, 32..63).
// D layout: lane = pixel, 8 VGPRs = 8 consecutive out-channels -> one b128
// bf16 store into channels-last h_pad.
// ---------------------------------------------------------------------------
__global__ __launch_bounds__(256) void conv1_kernel(
    const bf16_t* __restrict__ x_pad, const bf16_t* __restrict__ Wc,
    const float* __restrict__ b_conv, bf16_t* __restrict__ h_pad) {
  const int lane = threadIdx.x & 31;
  const int wave = threadIdx.x >> 5;   // out-channel tile 0..7
  const int lp   = lane & 15;
  const int half = lane >> 4;
  const int xg = blockIdx.x;           // 0..15
  const int py = blockIdx.y;           // 0..255
  const int b  = blockIdx.z;           // 0..3
  const int px = xg * 16 + lp;

  v8f acc = {};
  const bf16_t* wrow = Wc + (size_t)(wave * 16 + lp) * C_;
  for (int tap = 0; tap < 9; ++tap) {
    const int dy = tap / 3, dx = tap % 3;
    const bf16_t* bp = x_pad +
        (((size_t)b * HP_ + (py + dy)) * WP_ + (px + dx)) * C_;
    const bf16_t* ar = wrow + (size_t)tap * C2_ * C_;
    for (int kc = 0; kc < 2; ++kc) {
      v16bf a  = load_a_frag(ar, kc * 32, half);
      v16bf bb = load_b_frag(bp, kc * 32, half);
      acc = __builtin_amdgcn_wmma_f32_16x16x32_bf16(
          false, a, false, bb, (short)0, acc, false, false);
    }
  }
  const int ocb = wave * 16 + half * 8;
  v8bf o;
  for (int r = 0; r < 8; ++r)
    o[r] = (bf16_t)(acc[r] + b_conv[ocb + r]);
  bf16_t* dst = h_pad +
      (((size_t)b * HP_ + (py + 1)) * WP_ + (px + 1)) * C2_ + ocb;
  *(v8bf*)dst = o;
}

// ---------------------------------------------------------------------------
// stage 2 (fused): per wave = 16 pixels.
//  GEMM1: Wall[64,128] @ h[128,16]  (4 m-tiles x 4 k-steps = 16 WMMAs)
//         -> tanh -> per-o mean / unbiased var over T=16 (cross-half shfl)
//  GEMM2: AU 3x3 conv via Waue[9][16,128] @ h-halo (36 WMMAs) -> sigmoid
// ---------------------------------------------------------------------------
__global__ __launch_bounds__(256) void stage2_kernel(
    const bf16_t* __restrict__ h_pad, const bf16_t* __restrict__ Wall,
    const bf16_t* __restrict__ Waue, const float* __restrict__ b_out,
    const float* __restrict__ b_aue, const float* __restrict__ lms,
    float* __restrict__ out) {
  const int lane = threadIdx.x & 31;
  const int wave = threadIdx.x >> 5;
  const int lp   = lane & 15;
  const int half = lane >> 4;

  const int g = blockIdx.x * 8 + wave;        // pixel-group id, 0..16383
  const int gPerRow = W_ / 16;                // 16
  const int gPerImg = H_ * gPerRow;           // 4096
  const int b   = g / gPerImg;
  const int rem = g % gPerImg;
  const int py  = rem / gPerRow;
  const int px  = (rem % gPerRow) * 16 + lp;

  // ---- GEMM1: masked-sample matrix ----
  v16bf bfr[4];
  {
    const bf16_t* hp = h_pad +
        (((size_t)b * HP_ + (py + 1)) * WP_ + (px + 1)) * C2_;
    for (int k = 0; k < 4; ++k) bfr[k] = load_b_frag(hp, k * 32, half);
  }
  v8f acc[4] = {{}, {}, {}, {}};
  for (int mt = 0; mt < 4; ++mt) {
    const bf16_t* ar = Wall + (size_t)(mt * 16 + lp) * C2_;
    for (int k = 0; k < 4; ++k) {
      v16bf a = load_a_frag(ar, k * 32, half);
      acc[mt] = __builtin_amdgcn_wmma_f32_16x16x32_bf16(
          false, a, false, bfr[k], (short)0, acc[mt], false, false);
    }
  }

  // ---- tanh + per-o sum / sumsq over this lane's 8 t-samples per o ----
  float sum[4] = {0.f, 0.f, 0.f, 0.f};
  float sq[4]  = {0.f, 0.f, 0.f, 0.f};
  for (int mt = 0; mt < 4; ++mt)
    for (int r = 0; r < 8; ++r) {
      int row = mt * 16 + half * 8 + r;   // row = t*4 + o
      int o   = row & 3;
      float s = tanhf(acc[mt][r] + b_out[o]);
      sum[o] += s;
      sq[o]  += s * s;
    }
  // combine the two lanes holding the same pixel (l and l+16)
  for (int o = 0; o < 4; ++o) {
    sum[o] += __shfl_xor(sum[o], 16, 32);
    sq[o]  += __shfl_xor(sq[o], 16, 32);
  }

  const size_t plane = (size_t)H_ * W_;
  const size_t N4    = (size_t)B_ * 4 * plane;   // one output tensor
  const size_t opix  = (size_t)py * W_ + px;
  {
    const int o0 = half * 2;   // lanes<16 store o=0,1 ; lanes>=16 store o=2,3
    for (int j = 0; j < 2; ++j) {
      int o = o0 + j;
      float mu  = sum[o] * (1.0f / T_);
      float var = (sq[o] - sum[o] * sum[o] * (1.0f / T_)) * (1.0f / (T_ - 1));
      size_t oi = ((size_t)b * 4 + o) * plane + opix;
      out[N4 + oi]     = var;                   // EU
      out[2 * N4 + oi] = mu + lms[oi];          // mean
    }
  }

  // ---- GEMM2: AU 3x3 conv 128->4 (rows 4..15 of A are zero) ----
  v8f acc2 = {};
  for (int tap = 0; tap < 9; ++tap) {
    const int dy = tap / 3, dx = tap % 3;
    const bf16_t* hp = h_pad +
        (((size_t)b * HP_ + (py + dy)) * WP_ + (px + dx)) * C2_;
    const bf16_t* ar = Waue + (size_t)(tap * 16 + lp) * C2_;
    for (int k = 0; k < 4; ++k) {
      v16bf a  = load_a_frag(ar, k * 32, half);
      v16bf bb = load_b_frag(hp, k * 32, half);
      acc2 = __builtin_amdgcn_wmma_f32_16x16x32_bf16(
          false, a, false, bb, (short)0, acc2, false, false);
    }
  }
  if (half == 0) {           // rows 0..3 (= o) live in lanes 0..15, r=0..3
    for (int r = 0; r < 4; ++r) {
      float v  = acc2[r] + b_aue[r];
      float au = 1.0f / (1.0f + __expf(-v));
      out[((size_t)b * 4 + r) * plane + opix] = au;
    }
  }
}

// ---------------------------------------------------------------------------
extern "C" void kernel_launch(void* const* d_in, const int* in_sizes, int n_in,
                              void* d_out, int out_size, void* d_ws,
                              size_t ws_size, hipStream_t stream) {
  (void)in_sizes; (void)n_in; (void)out_size; (void)ws_size;
  const float* x      = (const float*)d_in[0];
  const float* lms    = (const float*)d_in[1];
  const float* W_conv = (const float*)d_in[2];
  const float* b_conv = (const float*)d_in[3];
  const float* W_out  = (const float*)d_in[4];
  const float* b_out  = (const float*)d_in[5];
  const float* W_aue  = (const float*)d_in[6];
  const float* b_aue  = (const float*)d_in[7];
  const float* masks  = (const float*)d_in[8];
  float* out = (float*)d_out;

  // workspace carve-up (bf16 elements)
  bf16_t* x_pad = (bf16_t*)d_ws;
  const size_t nxpad = (size_t)B_ * HP_ * WP_ * C_;
  bf16_t* h_pad = x_pad + nxpad;
  const size_t nhpad = (size_t)B_ * HP_ * WP_ * C2_;
  bf16_t* Wc   = h_pad + nhpad;
  bf16_t* Wall = Wc + 9 * C2_ * C_;
  bf16_t* Waue = Wall + 64 * C2_;

  // 1) zero padded activation buffers (halo must read as 0)
  {
    size_t n16 = (nxpad + nhpad) * sizeof(bf16_t) / 16;
    zero_kernel<<<4096, 256, 0, stream>>>((uint4*)d_ws, n16);
  }
  // 2) pack x -> channels-last bf16 padded
  pack_x_kernel<<<(B_ * H_ * W_ + 255) / 256, 256, 0, stream>>>(x, x_pad);
  // 3) pack weights (conv1, fused mask matrix, AU)
  {
    const int nw = 9 * C2_ * C_ + 64 * C2_ + 9 * 16 * C2_;
    prep_w_kernel<<<(nw + 255) / 256, 256, 0, stream>>>(
        W_conv, W_out, masks, W_aue, Wc, Wall, Waue);
  }
  // 4) stage 1 conv (WMMA implicit GEMM)
  conv1_kernel<<<dim3(W_ / 16, H_, B_), 256, 0, stream>>>(
      x_pad, Wc, b_conv, h_pad);
  // 5) fused samples/mean/var + AU conv
  stage2_kernel<<<(B_ * H_ * W_ / 16) / 8, 256, 0, stream>>>(
      h_pad, Wall, Waue, b_out, b_aue, lms, out);
}
